// GraphSAGENet_52398601012039
// MI455X (gfx1250) — matrix operand order repaired and checked by reading, not compiled, not run on hardware
//
#include <hip/hip_runtime.h>

#define N_NODES   50000
#define N_EDGES   800000
#define HID       128
#define N_LAYERS  3
#define N_CLASSES 6
#define NUM_GRAPHS 64
#define BN_EPS    1e-5f

typedef __attribute__((ext_vector_type(16))) __bf16        v16bf;
typedef __attribute__((ext_vector_type(8)))  float          v8f;
typedef __attribute__((ext_vector_type(8)))  unsigned int   v8u;
typedef __attribute__((ext_vector_type(4)))  unsigned int   v4u;
typedef __attribute__((ext_vector_type(4)))  unsigned short us4;

// ---------- helpers ----------

__device__ inline unsigned short f2bf(float f) {
    unsigned int u = __float_as_uint(f);
    u = (u + 0x7FFFu + ((u >> 16) & 1u)) >> 16;   // round-to-nearest-even
    return (unsigned short)u;
}

__device__ inline void atomAddF(float* p, float v) {
    // relaxed agent-scope fp32 add -> global_atomic_add_f32 (non-returning)
    __hip_atomic_fetch_add(p, v, __ATOMIC_RELAXED, __HIP_MEMORY_SCOPE_AGENT);
}

// A fragment: 16x32 bf16, row-major source, ISA-interleaved K pairs.
// lane m = lane&15 ; hi = lane>>4 ; VGPR0-3 = K (hi*8 + 0..7), VGPR4-7 = K (16 + hi*8 + 0..7)
__device__ inline v16bf fragA(const v4u* __restrict__ row, int kb, int hi) {
    v4u lo = row[kb + hi];       // dwords: K = k0 + hi*8 .. +7
    v4u h4 = row[kb + 2 + hi];   // dwords: K = k0 + 16 + hi*8 .. +7
    v8u u;
    u[0] = lo[0]; u[1] = lo[1]; u[2] = lo[2]; u[3] = lo[3];
    u[4] = h4[0]; u[5] = h4[1]; u[6] = h4[2]; u[7] = h4[3];
    return __builtin_bit_cast(v16bf, u);
}

// B fragment: 32x16 bf16 with B[k][n] = W[n][k]; lane n = lane&15, elements 0-7 -> K = k0 + hi*8 + e, 8-15 -> K = k0 + 16 + hi*8 + (e-8)
__device__ inline v16bf fragB(const v4u* __restrict__ row, int kb, int hi) {
    v4u lo = row[kb + hi];
    v4u h4 = row[kb + 2 + hi];
    v8u u;
    u[0] = lo[0]; u[1] = lo[1]; u[2] = lo[2]; u[3] = lo[3];
    u[4] = h4[0]; u[5] = h4[1]; u[6] = h4[2]; u[7] = h4[3];
    return __builtin_bit_cast(v16bf, u);
}

// ---------- one-time prep ----------

__global__ void convert_weights_k(const float* __restrict__ Wl, const float* __restrict__ Wr,
                                  unsigned short* __restrict__ wlB, unsigned short* __restrict__ wrB,
                                  int n) {
    int t = blockIdx.x * blockDim.x + threadIdx.x;
    if (t < n) { wlB[t] = f2bf(Wl[t]); wrB[t] = f2bf(Wr[t]); }
}

__global__ void degree_k(const int* __restrict__ dst, float* __restrict__ rdeg) {
    int e = blockIdx.x * blockDim.x + threadIdx.x;
    if (e < N_EDGES) atomAddF(&rdeg[dst[e]], 1.0f);
}

__global__ void rdeg_fin_k(float* __restrict__ rdeg) {
    int i = blockIdx.x * blockDim.x + threadIdx.x;
    if (i < N_NODES) rdeg[i] = 1.0f / fmaxf(rdeg[i], 1.0f);
}

__global__ void x_init_k(const float* __restrict__ xin, float* __restrict__ xf,
                         unsigned short* __restrict__ xb) {
    int t = blockIdx.x * blockDim.x + threadIdx.x;   // N_NODES*32 threads
    if (t >= N_NODES * 32) return;
    int node = t >> 5, lane = t & 31;
    size_t off = (size_t)node * HID + lane * 4;
    float4 v = *(const float4*)(xin + off);
    *(float4*)(xf + off) = v;
    us4 b = { f2bf(v.x), f2bf(v.y), f2bf(v.z), f2bf(v.w) };
    *(us4*)(xb + off) = b;
}

// ---------- per-layer kernels ----------

// one wave per edge; lane handles 4 features (float4 gather + 4 hw f32 atomics)
__global__ __launch_bounds__(256) void scatter_k(const float* __restrict__ xf,
                                                 const int* __restrict__ src,
                                                 const int* __restrict__ dst,
                                                 float* __restrict__ agg) {
    int t = blockIdx.x * blockDim.x + threadIdx.x;
    int e = t >> 5, lane = t & 31;
    if (e >= N_EDGES) return;
    int s = src[e], d = dst[e];
    float4 v = *(const float4*)(xf + (size_t)s * HID + lane * 4);
    float* o = agg + (size_t)d * HID + lane * 4;
    atomAddF(o + 0, v.x); atomAddF(o + 1, v.y);
    atomAddF(o + 2, v.z); atomAddF(o + 3, v.w);
}

__global__ void aggconv_k(const float* __restrict__ agg, const float* __restrict__ rdeg,
                          unsigned short* __restrict__ aggb) {
    int t = blockIdx.x * blockDim.x + threadIdx.x;   // N_NODES*32 threads
    if (t >= N_NODES * 32) return;
    int node = t >> 5, lane = t & 31;
    size_t off = (size_t)node * HID + lane * 4;
    float r = rdeg[node];
    float4 v = *(const float4*)(agg + off);
    us4 b = { f2bf(v.x * r), f2bf(v.y * r), f2bf(v.z * r), f2bf(v.w * r) };
    *(us4*)(aggb + off) = b;
}

// h = aggb @ Wl^T + bias + xb @ Wr^T ; also accumulate per-column sum / sumsq for BN.
// One block = one 16-row stripe; 8 waves = 8 column tiles of 16. K = 128 -> 4 steps x 2 WMMA.
__global__ __launch_bounds__(256) void gemm_bn_k(const unsigned short* __restrict__ aggb,
                                                 const unsigned short* __restrict__ xb,
                                                 const unsigned short* __restrict__ wl,
                                                 const unsigned short* __restrict__ wr,
                                                 const float* __restrict__ bias,
                                                 float* __restrict__ h,
                                                 float* __restrict__ colsum,
                                                 float* __restrict__ colsq) {
    const int lane = threadIdx.x & 31;
    const int wave = threadIdx.x >> 5;
    const int m0 = blockIdx.x * 16;      // 50000 / 16 = 3125 blocks, exact
    const int n0 = wave * 16;            // 8 waves cover HID = 128
    const int nl = lane & 15;
    const int hi = lane >> 4;

    const float b = bias[n0 + nl];
    v8f c;
#pragma unroll
    for (int v = 0; v < 8; ++v) c[v] = b;

    const v4u* aggRow = (const v4u*)(aggb + (size_t)(m0 + nl) * HID);
    const v4u* xRow   = (const v4u*)(xb   + (size_t)(m0 + nl) * HID);
    const v4u* wlRow  = (const v4u*)(wl   + (size_t)(n0 + nl) * HID);
    const v4u* wrRow  = (const v4u*)(wr   + (size_t)(n0 + nl) * HID);

#pragma unroll
    for (int k0 = 0; k0 < HID; k0 += 32) {
        const int kb = k0 >> 3;   // v4u (16B) index base within a 128-elem bf16 row
        v16bf aA = fragA(aggRow, kb, hi);
        v16bf bL = fragB(wlRow, kb, hi);
        c = __builtin_amdgcn_wmma_f32_16x16x32_bf16(false, aA, false, bL, (short)0, c, false, false);
        v16bf aX = fragA(xRow, kb, hi);
        v16bf bR = fragB(wrRow, kb, hi);
        c = __builtin_amdgcn_wmma_f32_16x16x32_bf16(false, aX, false, bR, (short)0, c, false, false);
    }

    float s = 0.f, s2 = 0.f;
#pragma unroll
    for (int v = 0; v < 8; ++v) {
        float d = c[v];
        h[(size_t)(m0 + v + 8 * hi) * HID + n0 + nl] = d;  // D layout: VGPR v -> row M = v + 8*hi
        s += d; s2 += d * d;
    }
    atomAddF(&colsum[n0 + nl], s);
    atomAddF(&colsq [n0 + nl], s2);
}

__global__ void bn_fin_k(const float* __restrict__ colsum, const float* __restrict__ colsq,
                         const float* __restrict__ gamma, const float* __restrict__ beta,
                         float* __restrict__ scale, float* __restrict__ shift) {
    int c = threadIdx.x;
    if (c >= HID) return;
    const float invN = 1.0f / (float)N_NODES;
    float mu  = colsum[c] * invN;
    float var = fmaxf(colsq[c] * invN - mu * mu, 0.0f);
    float sc  = gamma[c] * rsqrtf(var + BN_EPS);
    scale[c] = sc;
    shift[c] = beta[c] - mu * sc;
}

__global__ void normalize_k(const float* __restrict__ h, const float* __restrict__ scale,
                            const float* __restrict__ shift, float* __restrict__ xf,
                            unsigned short* __restrict__ xb) {
    int t = blockIdx.x * blockDim.x + threadIdx.x;   // N_NODES*32 threads
    if (t >= N_NODES * 32) return;
    int node = t >> 5, lane = t & 31;
    size_t off = (size_t)node * HID + lane * 4;
    float4 v  = *(const float4*)(h + off);
    float4 sc = *(const float4*)(scale + lane * 4);
    float4 sh = *(const float4*)(shift + lane * 4);
    float4 y;
    y.x = fmaxf(v.x * sc.x + sh.x, 0.f);
    y.y = fmaxf(v.y * sc.y + sh.y, 0.f);
    y.z = fmaxf(v.z * sc.z + sh.z, 0.f);
    y.w = fmaxf(v.w * sc.w + sh.w, 0.f);
    *(float4*)(xf + off) = y;
    us4 bv = { f2bf(y.x), f2bf(y.y), f2bf(y.z), f2bf(y.w) };
    *(us4*)(xb + off) = bv;
}

// ---------- pooling + head ----------

__global__ void cnt_k(const int* __restrict__ batch, float* __restrict__ cnt) {
    int i = blockIdx.x * blockDim.x + threadIdx.x;
    if (i < N_NODES) atomAddF(&cnt[batch[i]], 1.0f);
}

__global__ __launch_bounds__(256) void pool_k(const float* __restrict__ xf,
                                              const int* __restrict__ batch,
                                              float* __restrict__ pooled) {
    int t = blockIdx.x * blockDim.x + threadIdx.x;   // N_NODES*32 threads
    if (t >= N_NODES * 32) return;
    int node = t >> 5, lane = t & 31;
    int g = batch[node];
    float4 v = *(const float4*)(xf + (size_t)node * HID + lane * 4);
    float* o = pooled + (size_t)g * HID + lane * 4;
    atomAddF(o + 0, v.x); atomAddF(o + 1, v.y);
    atomAddF(o + 2, v.z); atomAddF(o + 3, v.w);
}

__global__ __launch_bounds__(256) void head_k(const float* __restrict__ pooled,
                                              const float* __restrict__ cnt,
                                              const float* __restrict__ W1, const float* __restrict__ b1,
                                              const float* __restrict__ W2, const float* __restrict__ b2,
                                              float* __restrict__ out) {
    __shared__ float h1[NUM_GRAPHS * 64];
    __shared__ float inv[NUM_GRAPHS];
    int t = threadIdx.x;
    if (t < NUM_GRAPHS) inv[t] = 1.0f / fmaxf(cnt[t], 1.0f);
    __syncthreads();
    for (int i = t; i < NUM_GRAPHS * 64; i += 256) {
        int g = i >> 6, j = i & 63;
        const float* pg = pooled + (size_t)g * HID;
        const float* w  = W1 + (size_t)j * HID;
        float iv = inv[g];
        float s = b1[j];
        for (int c = 0; c < HID; ++c) s += pg[c] * iv * w[c];
        h1[i] = fmaxf(s, 0.f);
    }
    __syncthreads();
    for (int i = t; i < NUM_GRAPHS * N_CLASSES; i += 256) {
        int g = i / N_CLASSES, j = i % N_CLASSES;
        const float* hg = h1 + g * 64;
        const float* w  = W2 + (size_t)j * 64;
        float s = b2[j];
        for (int k = 0; k < 64; ++k) s += hg[k] * w[k];
        out[i] = s;
    }
}

// ---------- launch ----------

extern "C" void kernel_launch(void* const* d_in, const int* in_sizes, int n_in,
                              void* d_out, int out_size, void* d_ws, size_t ws_size,
                              hipStream_t stream) {
    const float* x_in  = (const float*)d_in[0];
    const int*   ei    = (const int*)d_in[1];
    const int*   src   = ei;
    const int*   dst   = ei + N_EDGES;
    const int*   batch = (const int*)d_in[2];
    const float* Wl    = (const float*)d_in[3];
    const float* bl    = (const float*)d_in[4];
    const float* Wr    = (const float*)d_in[5];
    const float* gamma = (const float*)d_in[6];
    const float* beta  = (const float*)d_in[7];
    const float* W1    = (const float*)d_in[8];
    const float* b1    = (const float*)d_in[9];
    const float* W2    = (const float*)d_in[10];
    const float* b2    = (const float*)d_in[11];
    float* out = (float*)d_out;

    // workspace carve (256B aligned slices)
    char* p = (char*)d_ws;
    auto carve = [&](size_t bytes) -> char* {
        char* r = p;
        p += (bytes + 255) & ~(size_t)255;
        return r;
    };
    unsigned short* wlB  = (unsigned short*)carve((size_t)N_LAYERS * HID * HID * 2);
    unsigned short* wrB  = (unsigned short*)carve((size_t)N_LAYERS * HID * HID * 2);
    float*          rdeg = (float*)carve((size_t)N_NODES * 4);
    float*          xf   = (float*)carve((size_t)N_NODES * HID * 4);
    unsigned short* xb   = (unsigned short*)carve((size_t)N_NODES * HID * 2);
    float*          agg  = (float*)carve((size_t)N_NODES * HID * 4);  // reused as h after aggconv
    unsigned short* aggb = (unsigned short*)carve((size_t)N_NODES * HID * 2);
    float*          colsum = (float*)carve(HID * 4);
    float*          colsq  = (float*)carve(HID * 4);
    float*          scale  = (float*)carve(HID * 4);
    float*          shift  = (float*)carve(HID * 4);
    float*          pooled = (float*)carve((size_t)NUM_GRAPHS * HID * 4);
    float*          cnt    = (float*)carve((size_t)NUM_GRAPHS * 4);

    const int nW = N_LAYERS * HID * HID;                 // 49152
    const int nodeFeatThreads = N_NODES * 32;            // 1.6M
    const int nfBlocks = (nodeFeatThreads + 255) / 256;  // 6250

    // one-time prep
    convert_weights_k<<<(nW + 255) / 256, 256, 0, stream>>>(Wl, Wr, wlB, wrB, nW);
    hipMemsetAsync(rdeg, 0, (size_t)N_NODES * 4, stream);
    degree_k<<<(N_EDGES + 255) / 256, 256, 0, stream>>>(dst, rdeg);
    rdeg_fin_k<<<(N_NODES + 255) / 256, 256, 0, stream>>>(rdeg);
    x_init_k<<<nfBlocks, 256, 0, stream>>>(x_in, xf, xb);

    for (int l = 0; l < N_LAYERS; ++l) {
        hipMemsetAsync(agg, 0, (size_t)N_NODES * HID * 4, stream);
        hipMemsetAsync(colsum, 0, HID * 4, stream);
        hipMemsetAsync(colsq, 0, HID * 4, stream);

        scatter_k<<<(N_EDGES * 32) / 256, 256, 0, stream>>>(xf, src, dst, agg);
        aggconv_k<<<nfBlocks, 256, 0, stream>>>(agg, rdeg, aggb);

        gemm_bn_k<<<N_NODES / 16, 256, 0, stream>>>(aggb, xb,
                                                    wlB + (size_t)l * HID * HID,
                                                    wrB + (size_t)l * HID * HID,
                                                    bl + l * HID,
                                                    agg /* h aliases agg */,
                                                    colsum, colsq);
        bn_fin_k<<<1, 128, 0, stream>>>(colsum, colsq, gamma + l * HID, beta + l * HID, scale, shift);
        normalize_k<<<nfBlocks, 256, 0, stream>>>(agg, scale, shift, xf, xb);
    }

    hipMemsetAsync(pooled, 0, (size_t)NUM_GRAPHS * HID * 4, stream);
    hipMemsetAsync(cnt, 0, (size_t)NUM_GRAPHS * 4, stream);
    cnt_k<<<(N_NODES + 255) / 256, 256, 0, stream>>>(batch, cnt);
    pool_k<<<nfBlocks, 256, 0, stream>>>(xf, batch, pooled);
    head_k<<<1, 256, 0, stream>>>(pooled, cnt, W1, b1, W2, b2, out);
}
